// MHSARPB_73675868995691
// MI455X (gfx1250) — matrix operand order repaired and checked by reading, not compiled, not run on hardware
//
#include <hip/hip_runtime.h>
#include <hip/hip_bf16.h>

// MHSA with relative position bias, window 7x7 (N=49 padded to 64), DIM=1024,
// 16 heads x 64, B=2048 windows. All three matmul stages use
// v_wmma_f32_16x16x32_bf16 (bf16 inputs, fp32 accumulate); GEMM tiles are
// staged into LDS by the Tensor Data Mover (TENSOR_LOAD_TO_LDS + TENSORcnt)
// where possible.

typedef __bf16 bf16_t;
typedef __attribute__((ext_vector_type(16))) __bf16 v16bf;
typedef __attribute__((ext_vector_type(8)))  __bf16 bf16x8;
typedef __attribute__((ext_vector_type(8)))  float  v8f;
typedef __attribute__((ext_vector_type(4)))  unsigned int u32x4;
typedef __attribute__((ext_vector_type(8)))  int i32x8;
typedef __attribute__((ext_vector_type(4)))  int i32x4;

#define HEADS 16
#define NTOK 49
#define NPAD 64
#define HD 64
#define DIM 1024
#define NWIN 2048
#define MROWS (NWIN * NTOK)   // 100352

#if __has_builtin(__builtin_amdgcn_tensor_load_to_lds) && \
    __has_builtin(__builtin_amdgcn_s_wait_tensorcnt)
#define USE_TDM 1
#else
#define USE_TDM 0
#endif

// ---------------------------------------------------------------------------
#if USE_TDM
// Issue a TDM 2-D tile load: [tile_rows][tile_k] bf16 elements, global row
// stride `stride_elems`, into contiguous LDS at byte address `lds_addr`.
// D# packing per CDNA5 ISA section 8.3/8.4 (group0 128b, group1 256b).
__device__ __forceinline__ void tdm_load_tile_2d(unsigned lds_addr,
                                                 const void* gptr,
                                                 int tile_k, int tile_rows,
                                                 int stride_elems,
                                                 int tensor_d0, int tensor_d1) {
  unsigned long long ga = (unsigned long long)gptr;
  u32x4 g0;
  g0[0] = 1u;                                        // count=1 (valid user D#)
  g0[1] = lds_addr;                                  // lds_addr [63:32]
  g0[2] = (unsigned)(ga & 0xFFFFFFFFu);              // global_addr lo
  g0[3] = (unsigned)((ga >> 32) & 0x1FFFFFFu)        // global_addr hi (57b)
          | 0x80000000u;                             // type=2 ("image")
  i32x8 g1;
  g1[0] = 0x00010000;                                // data_size=1 -> 2 bytes
  g1[1] = (tensor_d0 & 0xFFFF) << 16;                // tensor_dim0[15:0]
  g1[2] = ((tensor_d0 >> 16) & 0xFFFF)               // tensor_dim0[31:16]
          | ((tensor_d1 & 0xFFFF) << 16);            // tensor_dim1[15:0]
  g1[3] = ((tensor_d1 >> 16) & 0xFFFF)               // tensor_dim1[31:16]
          | ((tile_k & 0xFFFF) << 16);               // tile_dim0
  g1[4] = (tile_rows & 0xFFFF);                      // tile_dim1 (tile_dim2=0)
  g1[5] = stride_elems;                              // tensor_dim0_stride lo32
  g1[6] = 0;                                         // stride hi / dim1_stride
  g1[7] = 0;
  i32x4 z4 = {0, 0, 0, 0};
#if defined(__clang_major__) && (__clang_major__ >= 23)
  i32x8 z8 = {0, 0, 0, 0, 0, 0, 0, 0};
  __builtin_amdgcn_tensor_load_to_lds(g0, g1, z4, z4, z8, 0);
#else
  __builtin_amdgcn_tensor_load_to_lds(g0, g1, z4, z4, 0);
#endif
}
#endif  // USE_TDM

// ---------------------------------------------------------------------------
// WMMA fragment loaders (per CDNA5 ISA 7.12.2, wave32).
// A (16x32 bf16, MxK): lanes 0-15 -> M=lane, K chunks [kb..kb+7] and
// [kb+16..kb+23] with kb = (lane>=16 ? 8 : 0).  Source must be K-contiguous.
__device__ __forceinline__ v16bf load_frag_A(const bf16_t* __restrict__ rowp,
                                             int k0, int lane) {
  int kb = k0 + ((lane >> 4) << 3);
  bf16x8 lo = *(const bf16x8*)(rowp + kb);
  bf16x8 hi = *(const bf16x8*)(rowp + kb + 16);
  v16bf f;
#pragma unroll
  for (int v = 0; v < 8; ++v) { f[v] = lo[v]; f[8 + v] = hi[v]; }
  return f;
}

// B (32x16 bf16, KxN): lane holds column n = lane&15, K chunks [kb..kb+7] and
// [kb+16..kb+23], kb = (lane>=16 ? 8 : 0).  Source (stored [n][k]) must be K-contiguous.
__device__ __forceinline__ v16bf load_frag_B(const bf16_t* __restrict__ colp,
                                             int k0, int lane) {
  int kb = k0 + ((lane >> 4) << 3);
  bf16x8 lo = *(const bf16x8*)(colp + kb);
  bf16x8 hi = *(const bf16x8*)(colp + kb + 16);
  v16bf f;
#pragma unroll
  for (int v = 0; v < 8; ++v) { f[v] = lo[v]; f[8 + v] = hi[v]; }
  return f;
}

// ---------------------------------------------------------------------------
// Tiled GEMM: C[M,N] = A[M,K] * W[N,K]^T + bias[N]
// A: fp32 (converted on the fly) or bf16, row-major [M][K].
// W: bf16 row-major [N][K] (torch weight layout == WMMA-B-friendly).
// Block tile 128x128, K-step 64, 256 threads = 8 waves (4 x 2), wave = 32x64.
// bf16 tiles are staged by the Tensor Data Mover; fp32 A is staged manually
// (TDM cannot convert fp32->bf16).
template <bool A_F32, bool OUT_BF16>
__global__ __launch_bounds__(256) void gemm_wmma_kernel(
    const void* __restrict__ Ap, const bf16_t* __restrict__ Wp,
    const float* __restrict__ bias, void* __restrict__ Cp,
    int M, int N, int K) {
  __shared__ __align__(16) bf16_t As[128 * 64];   // [m][k]
  __shared__ __align__(16) bf16_t Bs[128 * 64];   // [n][k]

  const int tid = threadIdx.x;
  const int lane = tid & 31, wave = tid >> 5;
  const int bm = blockIdx.y * 128, bn = blockIdx.x * 128;
  const int wm = (wave & 3) * 32, wn = (wave >> 2) * 64;

  v8f acc[2][4] = {};

  for (int k0 = 0; k0 < K; k0 += 64) {
#if USE_TDM
    // ---- TDM staging: one descriptor per tile, issued by thread 0 ----------
    if (tid == 0) {
      tdm_load_tile_2d((unsigned)(unsigned long long)(void*)Bs,
                       Wp + (size_t)bn * K + k0, 64, 128, K, K - k0, N - bn);
      if (!A_F32)
        tdm_load_tile_2d((unsigned)(unsigned long long)(void*)As,
                         (const bf16_t*)Ap + (size_t)bm * K + k0,
                         64, 128, K, K - k0, M - bm);
      __builtin_amdgcn_s_wait_tensorcnt(0);
    }
    if (A_F32) {
#pragma unroll
      for (int rep = 0; rep < 4; ++rep) {
        int idx = (tid + rep * 256) * 8;
        int r = idx >> 6, c = idx & 63;
        const float* Af = (const float*)Ap;
        const float4* g = (const float4*)(Af + (size_t)(bm + r) * K + k0 + c);
        float4 g0 = g[0], g1 = g[1];
        bf16x8 v;
        v[0] = (bf16_t)g0.x; v[1] = (bf16_t)g0.y;
        v[2] = (bf16_t)g0.z; v[3] = (bf16_t)g0.w;
        v[4] = (bf16_t)g1.x; v[5] = (bf16_t)g1.y;
        v[6] = (bf16_t)g1.z; v[7] = (bf16_t)g1.w;
        *(bf16x8*)(As + r * 64 + c) = v;
        if (k0 + 64 < K)
          __builtin_prefetch(Af + (size_t)(bm + r) * K + k0 + 64 + c, 0, 3);
      }
    }
    __syncthreads();
#else
    // ---- manual staging fallback ------------------------------------------
#pragma unroll
    for (int rep = 0; rep < 4; ++rep) {
      int idx = (tid + rep * 256) * 8;
      int r = idx >> 6, c = idx & 63;
      if (A_F32) {
        const float* Af = (const float*)Ap;
        const float4* g = (const float4*)(Af + (size_t)(bm + r) * K + k0 + c);
        float4 g0 = g[0], g1 = g[1];
        bf16x8 v;
        v[0] = (bf16_t)g0.x; v[1] = (bf16_t)g0.y;
        v[2] = (bf16_t)g0.z; v[3] = (bf16_t)g0.w;
        v[4] = (bf16_t)g1.x; v[5] = (bf16_t)g1.y;
        v[6] = (bf16_t)g1.z; v[7] = (bf16_t)g1.w;
        *(bf16x8*)(As + r * 64 + c) = v;
      } else {
        *(bf16x8*)(As + r * 64 + c) =
            *(const bf16x8*)((const bf16_t*)Ap + (size_t)(bm + r) * K + k0 + c);
      }
      *(bf16x8*)(Bs + r * 64 + c) =
          *(const bf16x8*)(Wp + (size_t)(bn + r) * K + k0 + c);
    }
    __syncthreads();
#endif

    // ---- 2 (M) x 4 (N) fragments, 2 inner K steps of 32 --------------------
#pragma unroll
    for (int kk = 0; kk < 64; kk += 32) {
      v16bf af[2], bfg[4];
#pragma unroll
      for (int i = 0; i < 2; ++i)
        af[i] = load_frag_A(As + (wm + i * 16 + (lane & 15)) * 64, kk, lane);
#pragma unroll
      for (int j = 0; j < 4; ++j)
        bfg[j] = load_frag_B(Bs + (wn + j * 16 + (lane & 15)) * 64, kk, lane);
#pragma unroll
      for (int i = 0; i < 2; ++i)
#pragma unroll
        for (int j = 0; j < 4; ++j)
          acc[i][j] = __builtin_amdgcn_wmma_f32_16x16x32_bf16(
              false, af[i], false, bfg[j], (short)0, acc[i][j], false, false);
    }
    __syncthreads();
  }

  // ---- epilogue: + bias, store ---------------------------------------------
#pragma unroll
  for (int i = 0; i < 2; ++i)
#pragma unroll
    for (int j = 0; j < 4; ++j)
#pragma unroll
      for (int r = 0; r < 8; ++r) {
        int row = bm + wm + i * 16 + r + ((lane >> 4) << 3);
        int col = bn + wn + j * 16 + (lane & 15);
        float v = acc[i][j][r] + bias[col];
        if (OUT_BF16)
          ((bf16_t*)Cp)[(size_t)row * N + col] = (bf16_t)v;
        else
          ((float*)Cp)[(size_t)row * N + col] = v;
      }
}

// ---------------------------------------------------------------------------
// fp32 -> bf16 elementwise (weights; layout kept as-is, [n][k])
__global__ void convert_bf16_kernel(const float* __restrict__ w,
                                    bf16_t* __restrict__ wb, long n) {
  long i = (long)blockIdx.x * 256 + threadIdx.x;
  if (i < n) wb[i] = (bf16_t)w[i];
}

// Padded relative-position bias: bp[h][64][64]; pad cols/rows get -1e30.
// Replicates _rpb_index incl. flipud/fliplr: idx[n,m] = raw[48-n, 48-m].
__global__ void build_bias_kernel(const float* __restrict__ rpb,
                                  float* __restrict__ bp) {
  int i = blockIdx.x * 256 + threadIdx.x;
  if (i >= HEADS * NPAD * NPAD) return;
  int h = i >> 12, n = (i >> 6) & 63, m = i & 63;
  float v = -1e30f;
  if (n < NTOK && m < NTOK) {
    int np = 48 - n, mp = 48 - m;
    int d0 = (np / 7) - (mp / 7) + 6;
    int d1 = (np % 7) - (mp % 7) + 6;
    v = rpb[h * 169 + d0 * 13 + d1];
  }
  bp[i] = v;
}

// ---------------------------------------------------------------------------
// Attention: one block per (window b, head h).  128 threads = 4 waves.
// S = (Q*scale) K^T + bias -> softmax -> P V, all WMMA on 64-padded tiles.
__global__ __launch_bounds__(128) void attn_win49_kernel(
    const bf16_t* __restrict__ qkv,     // [MROWS][3*DIM] bf16
    const float* __restrict__ biasP,    // [HEADS][64][64] fp32
    bf16_t* __restrict__ attnb) {       // [MROWS][DIM] bf16
  __shared__ __align__(16) bf16_t Qs[NPAD * HD];       // [n][d]
  __shared__ __align__(16) bf16_t Ks[NPAD * HD];       // [m][d] (K-contig in d)
  __shared__ __align__(16) bf16_t Vt[HD * NPAD];       // [d][m] (K-contig in m)
  __shared__ __align__(16) float  Sf[NPAD * 65];       // padded stride
  __shared__ __align__(16) bf16_t Pb[NPAD * NPAD];     // [n][m]

  const int b = blockIdx.x >> 4;
  const int h = blockIdx.x & 15;
  const int tid = threadIdx.x, lane = tid & 31, wave = tid >> 5;
  const size_t rowbase = (size_t)b * NTOK;

  // ---- load Q (scaled), K, V^T into LDS, zero-padding rows >= 49 ----------
  for (int it = tid; it < NPAD * (HD / 8); it += 128) {
    int r = it >> 3, c = (it & 7) << 3;
    bf16x8 q = {}, k = {};
    if (r < NTOK) {
      const bf16_t* p = qkv + (rowbase + r) * (3 * DIM) + h * HD + c;
      bf16x8 qg = *(const bf16x8*)(p);
      bf16x8 kg = *(const bf16x8*)(p + DIM);
      bf16x8 vg = *(const bf16x8*)(p + 2 * DIM);
#pragma unroll
      for (int e = 0; e < 8; ++e) {
        q[e] = (bf16_t)((float)qg[e] * 0.125f);   // scale = 64^-0.5
        k[e] = kg[e];
        Vt[(c + e) * NPAD + r] = vg[e];
      }
    } else {
#pragma unroll
      for (int e = 0; e < 8; ++e) Vt[(c + e) * NPAD + r] = (bf16_t)0.0f;
    }
    *(bf16x8*)(Qs + r * HD + c) = q;
    *(bf16x8*)(Ks + r * HD + c) = k;
  }
  __syncthreads();

  // ---- S = Q K^T ------------------------------------------------------------
  const int wrow = wave * 16;
  v8f sacc[4] = {};
#pragma unroll
  for (int kk = 0; kk < HD; kk += 32) {
    v16bf aq = load_frag_A(Qs + (wrow + (lane & 15)) * HD, kk, lane);
#pragma unroll
    for (int j = 0; j < 4; ++j) {
      v16bf bk = load_frag_B(Ks + (j * 16 + (lane & 15)) * HD, kk, lane);
      sacc[j] = __builtin_amdgcn_wmma_f32_16x16x32_bf16(
          false, aq, false, bk, (short)0, sacc[j], false, false);
    }
  }
  const float* bh = biasP + h * (NPAD * NPAD);
#pragma unroll
  for (int j = 0; j < 4; ++j)
#pragma unroll
    for (int r = 0; r < 8; ++r) {
      int row = wrow + r + ((lane >> 4) << 3);
      int col = j * 16 + (lane & 15);
      Sf[row * 65 + col] = sacc[j][r] + bh[row * NPAD + col];
    }
  __syncthreads();

  // ---- row softmax (pad cols carry -1e30 bias -> exp ~ 0) -------------------
  if (tid < NPAD) {
    float mx = -1e30f;
    for (int m = 0; m < NPAD; ++m) mx = fmaxf(mx, Sf[tid * 65 + m]);
    float s = 0.0f;
    for (int m = 0; m < NPAD; ++m) s += __expf(Sf[tid * 65 + m] - mx);
    float inv = 1.0f / s;
    for (int m = 0; m < NPAD; ++m)
      Pb[tid * NPAD + m] = (bf16_t)(__expf(Sf[tid * 65 + m] - mx) * inv);
  }
  __syncthreads();

  // ---- O = P V --------------------------------------------------------------
  v8f oacc[4] = {};
#pragma unroll
  for (int kk = 0; kk < NPAD; kk += 32) {
    v16bf ap = load_frag_A(Pb + (wrow + (lane & 15)) * NPAD, kk, lane);
#pragma unroll
    for (int j = 0; j < 4; ++j) {
      v16bf bv = load_frag_B(Vt + (j * 16 + (lane & 15)) * NPAD, kk, lane);
      oacc[j] = __builtin_amdgcn_wmma_f32_16x16x32_bf16(
          false, ap, false, bv, (short)0, oacc[j], false, false);
    }
  }
#pragma unroll
  for (int j = 0; j < 4; ++j)
#pragma unroll
    for (int r = 0; r < 8; ++r) {
      int row = wrow + r + ((lane >> 4) << 3);
      if (row < NTOK) {
        int d = j * 16 + (lane & 15);
        attnb[(rowbase + row) * DIM + h * HD + d] = (bf16_t)oacc[j][r];
      }
    }
}

// ---------------------------------------------------------------------------
extern "C" void kernel_launch(void* const* d_in, const int* in_sizes, int n_in,
                              void* d_out, int out_size, void* d_ws,
                              size_t ws_size, hipStream_t stream) {
  const float* x      = (const float*)d_in[0];  // [2048,7,7,1024]
  const float* qkv_w  = (const float*)d_in[1];  // [3072,1024]  ([n][k])
  const float* qkv_b  = (const float*)d_in[2];  // [3072]
  const float* rpb    = (const float*)d_in[3];  // [16,13,13]
  const float* proj_w = (const float*)d_in[4];  // [1024,1024]  ([n][k])
  const float* proj_b = (const float*)d_in[5];  // [1024]
  float* out = (float*)d_out;

  // workspace carve-up (256B aligned); total ~ 831 MB
  size_t off = 0;
  auto carve = [&](size_t bytes) {
    void* p = (char*)d_ws + off;
    off += (bytes + 255) & ~(size_t)255;
    return p;
  };
  bf16_t* wqkvB = (bf16_t*)carve((size_t)3 * DIM * DIM * 2);       // [3072][1024]
  bf16_t* projB = (bf16_t*)carve((size_t)DIM * DIM * 2);           // [1024][1024]
  float*  biasP = (float*)carve((size_t)HEADS * NPAD * NPAD * 4);
  bf16_t* qkv   = (bf16_t*)carve((size_t)MROWS * 3 * DIM * 2);     // 616 MB
  bf16_t* attnb = (bf16_t*)carve((size_t)MROWS * DIM * 2);         // 205 MB

  // 1) weights -> bf16 (layout preserved, already WMMA-B friendly [n][k])
  {
    long t1 = (long)3 * DIM * DIM;
    convert_bf16_kernel<<<(int)((t1 + 255) / 256), 256, 0, stream>>>(qkv_w, wqkvB, t1);
    long t2 = (long)DIM * DIM;
    convert_bf16_kernel<<<(int)((t2 + 255) / 256), 256, 0, stream>>>(proj_w, projB, t2);
  }
  // 2) padded RPB table
  build_bias_kernel<<<(HEADS * NPAD * NPAD) / 256, 256, 0, stream>>>(rpb, biasP);

  // 3) QKV GEMM: [100352,1024] fp32 x [3072,1024]^T bf16 -> bf16
  gemm_wmma_kernel<true, true><<<dim3(3 * DIM / 128, MROWS / 128), 256, 0, stream>>>(
      x, wqkvB, qkv_b, qkv, MROWS, 3 * DIM, DIM);

  // 4) attention per (window, head)
  attn_win49_kernel<<<NWIN * HEADS, 128, 0, stream>>>(qkv, biasP, attnb);

  // 5) output projection: [100352,1024] bf16 x [1024,1024]^T bf16 -> fp32 out
  gemm_wmma_kernel<false, false><<<dim3(DIM / 128, MROWS / 128), 256, 0, stream>>>(
      attnb, projB, proj_b, out, MROWS, DIM, DIM);
}
